// GAT_5403068858881
// MI455X (gfx1250) — matrix operand order repaired and checked by reading, not compiled, not run on hardware
//
#include <hip/hip_runtime.h>
#include <hip/hip_bf16.h>
#include <math.h>

// Problem constants (from the reference)
#define NN   100000
#define EE   1600000
#define INC  128
#define HIDC 32
#define GG   256
#define NEGS 0.2f

typedef __attribute__((ext_vector_type(16))) _Float16     v16h;
typedef __attribute__((ext_vector_type(8)))  float        v8f;
typedef __attribute__((ext_vector_type(4)))  unsigned int u32x4;

union F16Frag { v16h v; u32x4 q[2]; };

// ---------------------------------------------------------------- helpers
__device__ __forceinline__ void edge_sd(const int* __restrict__ ei, int e, int E,
                                        int& s, int& d) {
    if (e < E) { s = ei[e]; d = ei[E + e]; }
    else       { s = e - E; d = e - E; }      // self-loops appended after E
}

__device__ __forceinline__ float leaky(float v) {
    return v > 0.0f ? v : NEGS * v;
}

// sign-split float atomic max (init pattern 0xFFFFFFFF is identity for both orders)
__device__ __forceinline__ void atomicMaxF32(float* addr, float val) {
    if (val >= 0.0f) atomicMax((int*)addr,          __float_as_int(val));
    else             atomicMin((unsigned int*)addr, __float_as_uint(val));
}

// ---------------------------------------------------------------- conversions
__global__ void k_f32_to_f16(const float* __restrict__ src, _Float16* __restrict__ dst, int n) {
    int i = blockIdx.x * blockDim.x + threadIdx.x;
    if (i < n) dst[i] = (_Float16)src[i];
}

// Wt[n][k] = W[k][n]  (transpose so B columns are K-contiguous for the WMMA frag)
__global__ void k_build_wt(const float* __restrict__ W, _Float16* __restrict__ Wt,
                           int K, int Nout) {
    int i = blockIdx.x * blockDim.x + threadIdx.x;
    if (i >= K * Nout) return;
    int n = i / K, k = i % K;
    Wt[i] = (_Float16)W[k * Nout + n];
}

// ---------------------------------------------------------------- WMMA GEMM
// C[M x (16*NT)] = A[M x K] * Bt[(16*NT) x K]^T   (A,Bt f16 row-major, C f32)
// One wave32 per 16-row strip: the A fragment is loaded ONCE per 32-K step and
// reused across NT independent accumulators (8x less A traffic for NT=8), so the
// inner loop issues NT back-to-back v_wmma_f32_16x16x32_f16 per pair of A loads.
template <int NT>
__global__ void k_wmma_gemm_strip(const _Float16* __restrict__ A,
                                  const _Float16* __restrict__ Bt,
                                  float* __restrict__ C,
                                  int M, int K) {
    const int Nout = 16 * NT;
    int wave = (blockIdx.x * blockDim.x + threadIdx.x) >> 5;
    int lane = threadIdx.x & 31;
    if (wave >= (M >> 4)) return;                  // wave-uniform: EXEC stays all-ones

    int lo = lane & 15;     // row (A) / column (B)
    int hl = lane >> 4;     // half-wave select

    const _Float16* arow = A + (size_t)(wave * 16 + lo) * K;

    v8f acc[NT];
#pragma unroll
    for (int t = 0; t < NT; ++t) acc[t] = {};

    for (int kk = 0; kk < K; kk += 32) {
        // A 16x32 f16 frag: lane=row; halves j=0..7 -> K = kk+hl*8+j ; j=8..15 -> kk+16+hl*8+(j-8)
        F16Frag a;
        a.q[0] = *(const u32x4*)(arow + kk + hl * 8);
        a.q[1] = *(const u32x4*)(arow + kk + 16 + hl * 8);
#pragma unroll
        for (int t = 0; t < NT; ++t) {
            // B 32x16 f16 frag: lane=column; halves j=0..7 -> K = kk+hl*8+j ; j=8..15 -> kk+16+hl*8+(j-8)
            const _Float16* brow = Bt + (size_t)(t * 16 + lo) * K;
            F16Frag b;
            b.q[0] = *(const u32x4*)(brow + kk + hl * 8);
            b.q[1] = *(const u32x4*)(brow + kk + 16 + hl * 8);
            acc[t] = __builtin_amdgcn_wmma_f32_16x16x32_f16(
                false, a.v, false, b.v, (short)0, acc[t], false, false);
        }
    }

    // D layout: lane half selects M block of 8, VGPR r is row within block, lane%16 is N
#pragma unroll
    for (int t = 0; t < NT; ++t) {
        float* crow = C + (size_t)(wave * 16 + hl * 8) * Nout + t * 16 + lo;
#pragma unroll
        for (int r = 0; r < 8; ++r)
            crow[(size_t)r * Nout] = acc[t][r];
    }
}

// ---------------------------------------------------------------- attention
// asrc[n,h] = <h[n,h,:], a_s[h,:]>, adst likewise
__global__ void k_logits(const float* __restrict__ Hf,
                         const float* __restrict__ a_s, const float* __restrict__ a_d,
                         float* __restrict__ asrc, float* __restrict__ adst,
                         int n, int H, int C) {
    int idx = blockIdx.x * blockDim.x + threadIdx.x;
    if (idx >= n * H) return;
    int node = idx / H, hh = idx % H;
    const float* hp = Hf + (size_t)node * H * C + hh * C;
    const float* as = a_s + hh * C;
    const float* ad = a_d + hh * C;
    float s = 0.0f, d = 0.0f;
    for (int c = 0; c < C; ++c) { float v = hp[c]; s += v * as[c]; d += v * ad[c]; }
    asrc[idx] = s;
    adst[idx] = d;
}

__global__ void k_edge_max(const int* __restrict__ ei,
                           const float* __restrict__ asrc, const float* __restrict__ adst,
                           float* __restrict__ mx, int E, int Etot, int H) {
    int idx = blockIdx.x * blockDim.x + threadIdx.x;
    if (idx >= Etot * H) return;
    int e = idx / H, hh = idx % H;
    int s, d; edge_sd(ei, e, E, s, d);
    float v = leaky(asrc[s * H + hh] + adst[d * H + hh]);
    atomicMaxF32(&mx[d * H + hh], v);
}

__global__ void k_edge_den(const int* __restrict__ ei,
                           const float* __restrict__ asrc, const float* __restrict__ adst,
                           const float* __restrict__ mx, float* __restrict__ den,
                           int E, int Etot, int H) {
    int idx = blockIdx.x * blockDim.x + threadIdx.x;
    if (idx >= Etot * H) return;
    int e = idx / H, hh = idx % H;
    int s, d; edge_sd(ei, e, E, s, d);
    float v = leaky(asrc[s * H + hh] + adst[d * H + hh]);
    atomicAdd(&den[d * H + hh], expf(v - mx[d * H + hh]));
}

// weighted scatter-add; alpha recomputed (L2-resident operands) instead of materialized
__global__ void k_edge_agg(const int* __restrict__ ei,
                           const float* __restrict__ asrc, const float* __restrict__ adst,
                           const float* __restrict__ mx, const float* __restrict__ den,
                           const float* __restrict__ Hf, float* __restrict__ OutA,
                           int E, int Etot, int H, int C, int HC) {
    int idx = blockIdx.x * blockDim.x + threadIdx.x;
    if (idx >= Etot * HC) return;
    int e = idx / HC, c = idx % HC;
    int hh = c / C;
    int s, d; edge_sd(ei, e, E, s, d);
    float v = leaky(asrc[s * H + hh] + adst[d * H + hh]);
    float alpha = expf(v - mx[d * H + hh]) / (den[d * H + hh] + 1e-16f);
    atomicAdd(&OutA[(size_t)d * HC + c], alpha * Hf[(size_t)s * HC + c]);
}

// ---------------------------------------------------------------- activations
__global__ void k_elu_bias_f16(const float* __restrict__ in, const float* __restrict__ b,
                               _Float16* __restrict__ out, int n, int HC) {
    int i = blockIdx.x * blockDim.x + threadIdx.x;
    if (i >= n) return;
    float v = in[i] + b[i % HC];
    v = v > 0.0f ? v : expm1f(v);
    out[i] = (_Float16)v;
}

__global__ void k_elu_bias_f32(const float* __restrict__ in, const float* __restrict__ b,
                               float* __restrict__ out, int n, int HC) {
    int i = blockIdx.x * blockDim.x + threadIdx.x;
    if (i >= n) return;
    float v = in[i] + b[i % HC];
    out[i] = v > 0.0f ? v : expm1f(v);
}

// ---------------------------------------------------------------- pooling / readout
__global__ void k_count(const int* __restrict__ batch, float* __restrict__ counts, int n) {
    int i = blockIdx.x * blockDim.x + threadIdx.x;
    if (i < n) atomicAdd(&counts[batch[i]], 1.0f);
}

__global__ void k_pool(const int* __restrict__ batch, const float* __restrict__ H3,
                       float* __restrict__ pooled, int n, int C) {
    int i = blockIdx.x * blockDim.x + threadIdx.x;
    if (i >= n) return;
    int node = i / C, c = i % C;
    atomicAdd(&pooled[batch[node] * C + c], H3[i]);
}

__global__ void k_final(const float* __restrict__ pooled, const float* __restrict__ counts,
                        const float* __restrict__ fcw, const float* __restrict__ fcb,
                        float* __restrict__ out, int G, int C) {
    int g = blockIdx.x * blockDim.x + threadIdx.x;
    if (g >= G) return;
    float inv = 1.0f / fmaxf(counts[g], 1.0f);
    float acc = 0.0f;
    for (int c = 0; c < C; ++c) acc += pooled[g * C + c] * inv * fcw[c];
    out[g] = acc + fcb[0];
}

// ---------------------------------------------------------------- driver
extern "C" void kernel_launch(void* const* d_in, const int* in_sizes, int n_in,
                              void* d_out, int out_size, void* d_ws, size_t ws_size,
                              hipStream_t stream) {
    (void)in_sizes; (void)n_in; (void)out_size; (void)ws_size;

    const float* x    = (const float*)d_in[0];
    const int*   ei   = (const int*)d_in[1];
    const int*   bat  = (const int*)d_in[2];
    const float* W[3]  = {(const float*)d_in[3], (const float*)d_in[7],  (const float*)d_in[11]};
    const float* As[3] = {(const float*)d_in[4], (const float*)d_in[8],  (const float*)d_in[12]};
    const float* Ad[3] = {(const float*)d_in[5], (const float*)d_in[9],  (const float*)d_in[13]};
    const float* Bb[3] = {(const float*)d_in[6], (const float*)d_in[10], (const float*)d_in[14]};
    const float* fcw = (const float*)d_in[15];
    const float* fcb = (const float*)d_in[16];
    float* out = (float*)d_out;

    // bump allocator over workspace (hipMalloc base is 256B aligned)
    char* ws = (char*)d_ws;
    size_t off = 0;
    auto alloc = [&](size_t bytes) -> void* {
        void* p = ws + off;
        off += (bytes + 255) & ~(size_t)255;
        return p;
    };
    _Float16* Xh  = (_Float16*)alloc((size_t)NN * 128 * sizeof(_Float16));
    _Float16* Wt  = (_Float16*)alloc((size_t)128 * 128 * sizeof(_Float16));
    float* Hf     = (float*)alloc((size_t)NN * 128 * sizeof(float));
    float* OutA   = (float*)alloc((size_t)NN * 128 * sizeof(float));
    float* asrc   = (float*)alloc((size_t)NN * 4 * sizeof(float));
    float* adst   = (float*)alloc((size_t)NN * 4 * sizeof(float));
    float* mx     = (float*)alloc((size_t)NN * 4 * sizeof(float));
    float* den    = (float*)alloc((size_t)NN * 4 * sizeof(float));
    float* H3     = (float*)alloc((size_t)NN * HIDC * sizeof(float));
    float* pooled = (float*)alloc((size_t)GG * HIDC * sizeof(float));
    float* counts = (float*)alloc((size_t)GG * sizeof(float));

    const int Etot = EE + NN;
    const int TPB = 256;
    auto nb = [](long long n, int t) { return (unsigned)((n + t - 1) / t); };

    for (int L = 0; L < 3; ++L) {
        const int K  = 128;
        const int H  = (L == 2) ? 1 : 4;
        const int C  = HIDC;
        const int HC = H * C;

        if (L == 0)
            k_f32_to_f16<<<nb((long long)NN * 128, TPB), TPB, 0, stream>>>(x, Xh, NN * 128);
        // layers 1/2: Xh already produced by previous k_elu_bias_f16

        k_build_wt<<<nb((long long)K * HC, TPB), TPB, 0, stream>>>(W[L], Wt, K, HC);

        long long waves = (long long)(NN / 16);   // one wave per 16-row strip
        if (L < 2)
            k_wmma_gemm_strip<8><<<nb(waves * 32, TPB), TPB, 0, stream>>>(Xh, Wt, Hf, NN, K);
        else
            k_wmma_gemm_strip<2><<<nb(waves * 32, TPB), TPB, 0, stream>>>(Xh, Wt, Hf, NN, K);

        k_logits<<<nb((long long)NN * H, TPB), TPB, 0, stream>>>(Hf, As[L], Ad[L],
                                                                 asrc, adst, NN, H, C);

        hipMemsetAsync(mx,   0xFF, (size_t)NN * H * sizeof(float), stream); // -max for both orderings
        hipMemsetAsync(den,  0,    (size_t)NN * H * sizeof(float), stream);
        hipMemsetAsync(OutA, 0,    (size_t)NN * HC * sizeof(float), stream);

        k_edge_max<<<nb((long long)Etot * H, TPB), TPB, 0, stream>>>(ei, asrc, adst, mx,
                                                                     EE, Etot, H);
        k_edge_den<<<nb((long long)Etot * H, TPB), TPB, 0, stream>>>(ei, asrc, adst, mx, den,
                                                                     EE, Etot, H);
        k_edge_agg<<<nb((long long)Etot * HC, TPB), TPB, 0, stream>>>(ei, asrc, adst, mx, den,
                                                                      Hf, OutA, EE, Etot,
                                                                      H, C, HC);

        if (L < 2)
            k_elu_bias_f16<<<nb((long long)NN * HC, TPB), TPB, 0, stream>>>(OutA, Bb[L], Xh,
                                                                            NN * HC, HC);
        else
            k_elu_bias_f32<<<nb((long long)NN * HC, TPB), TPB, 0, stream>>>(OutA, Bb[L], H3,
                                                                            NN * HC, HC);
    }

    hipMemsetAsync(pooled, 0, (size_t)GG * HIDC * sizeof(float), stream);
    hipMemsetAsync(counts, 0, (size_t)GG * sizeof(float), stream);
    k_count<<<nb(NN, TPB), TPB, 0, stream>>>(bat, counts, NN);
    k_pool<<<nb((long long)NN * HIDC, TPB), TPB, 0, stream>>>(bat, H3, pooled, NN * HIDC, HIDC);
    k_final<<<1, TPB, 0, stream>>>(pooled, counts, fcw, fcb, out, GG, HIDC);
}
